// DecoderSemantic_11390253269670
// MI455X (gfx1250) — matrix operand order repaired and checked by reading, not compiled, not run on hardware
//
#include <hip/hip_runtime.h>

typedef __bf16 bf16x16 __attribute__((ext_vector_type(16)));
typedef __bf16 bf16x8v __attribute__((ext_vector_type(8)));
typedef float  f32x8   __attribute__((ext_vector_type(8)));
typedef unsigned short u16x8 __attribute__((ext_vector_type(8)));
typedef int i32x4 __attribute__((ext_vector_type(4)));
typedef int i32x8 __attribute__((ext_vector_type(8)));

union BF8U  { bf16x8v v; u16x8 u; };
union BF16U { bf16x16 v; bf16x8v h[2]; };

__device__ inline unsigned short f2bfbits(float f) {
  union { float f; unsigned u; } v; v.f = f;
  return (unsigned short)((v.u + 0x7FFFu + ((v.u >> 16) & 1u)) >> 16);
}
__device__ inline __bf16 f2bf(float f) {
  unsigned short h = f2bfbits(f);
  __bf16 r; __builtin_memcpy(&r, &h, 2); return r;
}
__device__ inline float bf2f(unsigned short h) {
  unsigned u = (unsigned)h << 16;
  float f; __builtin_memcpy(&f, &u, 4); return f;
}
__device__ inline bf16x8v ld8(const void* p) {   // 8 contig bf16, 16B aligned
  BF8U t; t.u = *(const u16x8*)p; return t.v;
}
__device__ inline bf16x8v zero8() {
  BF8U t; t.u = (u16x8)0; return t.v;
}

// TDM: pull a 16 x Ktot bf16 weight panel (row stride Ktot) into LDS.
// Issued by wave 0 only; caller does s_wait_tensorcnt + __syncthreads.
__device__ inline void tdm_panel_load(unsigned short* As, const __bf16* gsrc, int Ktot) {
  unsigned long long ga = (unsigned long long)(const void*)gsrc;
  unsigned lds_off = (unsigned)(size_t)(const void*)As;  // low 32 flat bits = LDS addr
  i32x4 g0;
  g0[0] = 1;                                    // count=1, user D#
  g0[1] = (int)lds_off;                         // lds_addr
  g0[2] = (int)(unsigned)(ga & 0xffffffffull);  // global_addr[31:0]
  g0[3] = (int)(((ga >> 32) & 0x1ffffffull) | 0x80000000ull);  // addr[56:32] | type=2
  i32x8 g1;
  g1[0] = 1 << 16;                              // data_size=1 (2 bytes)
  g1[1] = (int)(((unsigned)Ktot & 0xffffu) << 16);      // tensor_dim0 lo
  g1[2] = (int)(((unsigned)Ktot >> 16) | (16u << 16));  // tensor_dim0 hi | tensor_dim1
  g1[3] = (int)((unsigned)Ktot << 16);          // tensor_dim1 hi=0 | tile_dim0
  g1[4] = 16;                                   // tile_dim1=16, tile_dim2=0
  g1[5] = Ktot;                                 // tensor_dim0_stride lo
  g1[6] = 0;
  g1[7] = 0;
  asm volatile("tensor_load_to_lds %0, %1" : : "s"(g0), "s"(g1) : "memory");
}

// ---------------------------------------------------------------------------
// 3x3x3 SAME conv, implicit GEMM, Cin % 32 == 0 fast path.
// Tap-outer (kd,kh,kw) loops: no divisions in the hot path; bounds check and
// base address hoisted per tap. Inner body: 2x ds_load_b128 (A panel) +
// 2x global_load_b128 (B) + 1x v_wmma_f32_16x16x32_bf16.
// ---------------------------------------------------------------------------
__global__ __launch_bounds__(256) void conv3_tap_wmma(
    const __bf16* __restrict__ in, const __bf16* __restrict__ wb,
    const float* __restrict__ bn, __bf16* __restrict__ out_bf,
    float* __restrict__ out_f32,
    int Nb, int Cin, int lc, int Cout, int D, int H, int W) {
  extern __shared__ unsigned short As[];
  const int Ktot = 27 * Cin;
  const int tid = threadIdx.x;
  const int lane = tid & 31;
  const int wave = tid >> 5;
  const int khalf = lane >> 4;
  const int co_base = blockIdx.y << 4;
  const int HW = H * W, DHW = D * HW;
  const int total = Nb * DHW;

  if (wave == 0) {
    tdm_panel_load(As, wb + (size_t)co_base * Ktot, Ktot);
    __builtin_amdgcn_s_wait_tensorcnt(0);
  }
  __syncthreads();

  const int j = lane & 15;
  const int idx = (int)blockIdx.x * 128 + wave * 16 + j;
  const bool valid = idx < total;
  const int cidx = valid ? idx : 0;
  const int n = cidx / DHW;
  const int rem = cidx - n * DHW;
  const int d = rem / HW;
  const int r2 = rem - d * HW;
  const int h = r2 / W;
  const int w = r2 - h * W;

  const unsigned short* ap = As + (size_t)(lane & 15) * Ktot + khalf * 16;
  const int cch = Cin >> 5;       // ci chunks of 32
  f32x8 acc = {};
  int koff = 0;                   // tap * Cin
  for (int kd = -1; kd <= 1; ++kd) {
    const int dd = d + kd;
    const bool okd = (unsigned)dd < (unsigned)D;
    for (int kh = -1; kh <= 1; ++kh) {
      const int hh = h + kh;
      const bool okh = okd && (unsigned)hh < (unsigned)H;
      for (int kw = -1; kw <= 1; ++kw) {
        const int ww = w + kw;
        const bool inb = okh && (unsigned)ww < (unsigned)W;
        const __bf16* bp =
            in + ((size_t)(((n * D + dd) * H + hh) * W + ww) << lc) + khalf * 16;
        for (int cc = 0; cc < cch; ++cc) {
          BF16U a, b;
          const int ko = koff + (cc << 5);
          a.h[0] = ld8(ap + ko);
          a.h[1] = ld8(ap + ko + 8);
          if (inb) {
            b.h[0] = ld8(bp + (cc << 5));
            b.h[1] = ld8(bp + (cc << 5) + 8);
          } else {
            b.h[0] = zero8();
            b.h[1] = zero8();
          }
          acc = __builtin_amdgcn_wmma_f32_16x16x32_bf16(
              false, a.v, false, b.v, (short)0, acc, false, false);
        }
        koff += Cin;
      }
    }
  }

  if (valid) {
    float y[8];
    #pragma unroll
    for (int r = 0; r < 8; ++r) {
      int co = co_base + (khalf << 3) + r;
      float g = bn[co], be = bn[Cout + co];
      float m = bn[2 * Cout + co], va = bn[3 * Cout + co];
      float sc = g * rsqrtf(va + 1e-5f);
      y[r] = fmaxf(acc[r] * sc + (be - sc * m), 0.0f);
    }
    if (out_bf) {
      BF8U ov;
      #pragma unroll
      for (int r = 0; r < 8; ++r) ov.u[r] = f2bfbits(y[r]);
      *(u16x8*)(out_bf + (size_t)cidx * Cout + co_base + khalf * 8) = ov.u;
    }
    if (out_f32) {
      #pragma unroll
      for (int r = 0; r < 8; ++r) {
        int co = co_base + (khalf << 3) + r;
        out_f32[(size_t)(n * Cout + co) * DHW + rem] = y[r];
      }
    }
  }
}

// ---------------------------------------------------------------------------
// Generic chunked variant (handles Cin=16 / ragged K). Same structure as
// round 2; used only for the stage-2 conv2 launch.
// ---------------------------------------------------------------------------
__global__ __launch_bounds__(256) void conv3_wmma_nhwc(
    const __bf16* __restrict__ in, const __bf16* __restrict__ wb,
    const float* __restrict__ bn, __bf16* __restrict__ out_bf,
    float* __restrict__ out_f32,
    int Nb, int Cin, int lc, int Cout, int D, int H, int W) {
  extern __shared__ unsigned short As[];
  const int Ktot = 27 * Cin;
  const int tid = threadIdx.x;
  const int lane = tid & 31;
  const int wave = tid >> 5;
  const int khalf = lane >> 4;
  const int co_base = blockIdx.y << 4;
  const int HW = H * W, DHW = D * HW;
  const int total = Nb * DHW;
  const int cmask = Cin - 1;

  if (wave == 0) {
    As[16 * Ktot + lane] = 0;  // ragged-K A reads see finite bits
    tdm_panel_load(As, wb + (size_t)co_base * Ktot, Ktot);
    __builtin_amdgcn_s_wait_tensorcnt(0);
  }
  __syncthreads();

  const int j = lane & 15;
  const int idx = (int)blockIdx.x * 128 + wave * 16 + j;
  const bool valid = idx < total;
  const int cidx = valid ? idx : 0;
  const int n = cidx / DHW;
  const int rem = cidx - n * DHW;
  const int d = rem / HW;
  const int r2 = rem - d * HW;
  const int h = r2 / W;
  const int w = r2 - h * W;

  const unsigned short* arow = As + (size_t)(lane & 15) * Ktot;
  f32x8 acc = {};
  const int nch = (Ktot + 31) >> 5;
  for (int kc = 0; kc < nch; ++kc) {
    const int K0 = kc << 5;
    BF16U a, b;
    a.h[0] = ld8(arow + K0 + khalf * 16);
    a.h[1] = ld8(arow + K0 + khalf * 16 + 8);
    #pragma unroll
    for (int run = 0; run < 2; ++run) {
      int k8 = K0 + khalf * 16 + run * 8;
      int tap = k8 >> lc;
      bf16x8v vv = zero8();
      if (tap < 27) {
        int ci = k8 & cmask;
        int kd = tap / 9;
        int rr = tap - kd * 9;
        int kh = rr / 3;
        int kw = rr - kh * 3;
        int dd = d + kd - 1, hh = h + kh - 1, ww = w + kw - 1;
        if ((unsigned)dd < (unsigned)D && (unsigned)hh < (unsigned)H &&
            (unsigned)ww < (unsigned)W)
          vv = ld8(in + ((size_t)(((n * D + dd) * H + hh) * W + ww) << lc) + ci);
      }
      b.h[run] = vv;
    }
    acc = __builtin_amdgcn_wmma_f32_16x16x32_bf16(
        false, a.v, false, b.v, (short)0, acc, false, false);
  }

  if (valid) {
    float y[8];
    #pragma unroll
    for (int r = 0; r < 8; ++r) {
      int co = co_base + (khalf << 3) + r;
      float g = bn[co], be = bn[Cout + co];
      float m = bn[2 * Cout + co], va = bn[3 * Cout + co];
      float sc = g * rsqrtf(va + 1e-5f);
      y[r] = fmaxf(acc[r] * sc + (be - sc * m), 0.0f);
    }
    if (out_bf) {
      BF8U ov;
      #pragma unroll
      for (int r = 0; r < 8; ++r) ov.u[r] = f2bfbits(y[r]);
      *(u16x8*)(out_bf + (size_t)cidx * Cout + co_base + khalf * 8) = ov.u;
    }
    if (out_f32) {
      #pragma unroll
      for (int r = 0; r < 8; ++r) {
        int co = co_base + (khalf << 3) + r;
        out_f32[(size_t)(n * Cout + co) * DHW + rem] = y[r];
      }
    }
  }
}

// ---------------------------------------------------------------------------
// Transposed conv k2 s2: per-octant GEMM over Cin; weights pre-repacked and
// tap-flipped (JAX conv_transpose) to (8, Cout, Cin) bf16. All loads b128.
// ---------------------------------------------------------------------------
__global__ __launch_bounds__(32) void upconv_wmma_nhwc(
    const __bf16* __restrict__ in, const __bf16* __restrict__ uw,
    const float* __restrict__ bn, __bf16* __restrict__ cat,
    int Nb, int Cin, int Cout, int Di, int Hi, int Wi, int Ctot) {
  const int lane = threadIdx.x;
  const int khalf = lane >> 4;
  const int j = lane & 15;
  const int Si = Di * Hi * Wi;
  const int total = Nb * Si;
  const int oct = blockIdx.y;
  const int od = (oct >> 2) & 1, oh = (oct >> 1) & 1, ow = oct & 1;
  const int co_base = blockIdx.z << 4;

  const int idx = (int)blockIdx.x * 16 + j;
  const bool valid = idx < total;
  const int cidx = valid ? idx : 0;
  const int n = cidx / Si;
  const int p = cidx - n * Si;
  const int di = p / (Hi * Wi);
  const int pr = p - di * Hi * Wi;
  const int hi = pr / Wi;
  const int wi = pr - hi * Wi;

  const __bf16* arow = uw + ((size_t)oct * Cout + co_base + (lane & 15)) * Cin;
  const __bf16* brow = in + (size_t)cidx * Cin;

  f32x8 acc = {};
  const int nch = Cin >> 5;
  for (int kc = 0; kc < nch; ++kc) {
    const int K0 = kc << 5;
    BF16U a, b;
    a.h[0] = ld8(arow + K0 + khalf * 16);
    a.h[1] = ld8(arow + K0 + khalf * 16 + 8);
    b.h[0] = ld8(brow + K0 + khalf * 16);
    b.h[1] = ld8(brow + K0 + khalf * 16 + 8);
    acc = __builtin_amdgcn_wmma_f32_16x16x32_bf16(
        false, a.v, false, b.v, (short)0, acc, false, false);
  }

  if (valid) {
    const int Do = Di * 2, Ho = Hi * 2, Wo = Wi * 2;
    const size_t po = (size_t)((n * Do + 2 * di + od) * Ho + 2 * hi + oh) * Wo +
                      (2 * wi + ow);
    BF8U ov;
    #pragma unroll
    for (int r = 0; r < 8; ++r) {
      int co = co_base + (khalf << 3) + r;
      float g = bn[co], be = bn[Cout + co];
      float m = bn[2 * Cout + co], va = bn[3 * Cout + co];
      float sc = g * rsqrtf(va + 1e-5f);
      ov.u[r] = f2bfbits(fmaxf(acc[r] * sc + (be - sc * m), 0.0f));
    }
    *(u16x8*)(cat + po * Ctot + co_base + khalf * 8) = ov.u;
  }
}

// conv weights (Cout,Cin,3,3,3) f32 -> (Cout, 27*Cin) bf16, K = tap*Cin+ci
__global__ void conv_w_repack(const float* __restrict__ src,
                              __bf16* __restrict__ dst, int Cout, int Cin, int lc) {
  int i = blockIdx.x * 256 + threadIdx.x;
  int Ktot = 27 * Cin;
  if (i >= Cout * Ktot) return;
  int co = i / Ktot;
  int k = i - co * Ktot;
  int tap = k >> lc;
  int ci = k & (Cin - 1);
  dst[i] = f2bf(src[(co * Cin + ci) * 27 + tap]);
}

// upconv weights (Cin,Cout,2,2,2) f32 -> (8, Cout, Cin) bf16, tap-flipped
__global__ void up_w_repack(const float* __restrict__ src,
                            __bf16* __restrict__ dst, int Cout, int Cin) {
  int i = blockIdx.x * 256 + threadIdx.x;
  if (i >= 8 * Cout * Cin) return;
  int oct = i / (Cout * Cin);
  int r = i - oct * Cout * Cin;
  int co = r / Cin;
  int ci = r - co * Cin;
  dst[i] = f2bf(src[(ci * Cout + co) * 8 + (7 - oct)]);
}

// f32 NCDHW -> bf16 channels-last slot [Coff, Coff+C) of a Ctot-wide buffer
__global__ void act_to_nhwc(const float* __restrict__ src, __bf16* __restrict__ dst,
                            int Nb, int C, int S, int Ctot, int Coff) {
  int i = blockIdx.x * 256 + threadIdx.x;
  if (i >= Nb * C * S) return;
  int n = i / (C * S);
  int r = i - n * C * S;
  int c = r / S;
  int p = r - c * S;
  dst[((size_t)(n * S + p)) * Ctot + Coff + c] = f2bf(src[i]);
}

// 1x1x1 head: bf16 NDHWC (Cin=16) -> f32 NCDHW, Cls=20
__global__ void head_conv_kernel(const __bf16* __restrict__ in,
                                 const float* __restrict__ wgt,
                                 float* __restrict__ out, int Nb, int Cls, int S) {
  int t = blockIdx.x * 256 + threadIdx.x;
  if (t >= Nb * S) return;
  int n = t / S;
  int p = t - n * S;
  union { bf16x8v v; unsigned short s[8]; } a, b;
  a.v = ld8(in + (size_t)t * 16);
  b.v = ld8(in + (size_t)t * 16 + 8);
  float vin[16];
  #pragma unroll
  for (int ci = 0; ci < 8; ++ci) { vin[ci] = bf2f(a.s[ci]); vin[ci + 8] = bf2f(b.s[ci]); }
  for (int cls = 0; cls < Cls; ++cls) {
    float accv = 0.0f;
    #pragma unroll
    for (int ci = 0; ci < 16; ++ci) accv += wgt[cls * 16 + ci] * vin[ci];
    out[(size_t)(n * Cls + cls) * S + p] = accv;
  }
}

extern "C" void kernel_launch(void* const* d_in, const int* in_sizes, int n_in,
                              void* d_out, int out_size, void* d_ws, size_t ws_size,
                              hipStream_t stream) {
  (void)in_sizes; (void)n_in; (void)out_size; (void)ws_size;
  const float* Bx    = (const float*)d_in[0];
  const float* SK[3] = {(const float*)d_in[1], (const float*)d_in[2], (const float*)d_in[3]};
  const float* UW[3] = {(const float*)d_in[4], (const float*)d_in[10], (const float*)d_in[16]};
  const float* UB[3] = {(const float*)d_in[5], (const float*)d_in[11], (const float*)d_in[17]};
  const float* W1[3] = {(const float*)d_in[6], (const float*)d_in[12], (const float*)d_in[18]};
  const float* B1[3] = {(const float*)d_in[7], (const float*)d_in[13], (const float*)d_in[19]};
  const float* W2[3] = {(const float*)d_in[8], (const float*)d_in[14], (const float*)d_in[20]};
  const float* B2[3] = {(const float*)d_in[9], (const float*)d_in[15], (const float*)d_in[21]};
  const float* OW    = (const float*)d_in[22];

  float* outp = (float*)d_out;                      // (2,20,64^3) f32 NCDHW
  float* ups0 = outp + 2 * 20 * 64 * 64 * 64;
  float* ups1 = ups0 + 2 * 64 * 16 * 16 * 16;
  float* ups2 = ups1 + 2 * 32 * 32 * 32 * 32;
  float* Uf[3] = {ups0, ups1, ups2};

  // bf16 workspace carve-out (all offsets 16B aligned)
  __bf16* wsb = (__bf16*)d_ws;
  size_t o = 0;
  __bf16* xb   = wsb + o; o += 131072;      // b  -> (2,8^3,128)
  __bf16* catb = wsb + o; o += 16777216;    // concat, worst (2,64^3,32)
  __bf16* tbuf = wsb + o; o += 8388608;     // conv1 out, worst (2,64^3,16)
  __bf16* ubuf = wsb + o; o += 8388608;     // conv2 out, worst (2,64^3,16)
  __bf16* wb1[3]; __bf16* wb2[3]; __bf16* uwb[3];
  int chs[3] = {64, 32, 16};
  for (int s = 0; s < 3; ++s) {
    int ch = chs[s];
    wb1[s] = wsb + o; o += (size_t)ch * 27 * (2 * ch);
    wb2[s] = wsb + o; o += (size_t)ch * 27 * ch + 32;  // +32 ragged pad room
    uwb[s] = wsb + o; o += (size_t)8 * ch * (2 * ch);
  }

  const int Nb = 2;
  for (int s = 0; s < 3; ++s) {
    int ch = chs[s];
    int lc1 = 31 - __builtin_clz((unsigned)(2 * ch));
    int t1 = ch * 27 * 2 * ch, t2 = ch * 27 * ch, t3 = 8 * ch * 2 * ch;
    conv_w_repack<<<(t1 + 255) / 256, 256, 0, stream>>>(W1[s], wb1[s], ch, 2 * ch, lc1);
    conv_w_repack<<<(t2 + 255) / 256, 256, 0, stream>>>(W2[s], wb2[s], ch, ch, lc1 - 1);
    up_w_repack<<<(t3 + 255) / 256, 256, 0, stream>>>(UW[s], uwb[s], ch, 2 * ch);
  }
  act_to_nhwc<<<(2 * 128 * 512 + 255) / 256, 256, 0, stream>>>(Bx, xb, Nb, 128, 512, 128, 0);

  const __bf16* xbf = xb;
  int Di = 8;
  for (int s = 0; s < 3; ++s) {
    const int ch = chs[s];
    const int CinUp = 2 * ch;
    const int D = 2 * Di;
    const int Si = Di * Di * Di;
    const int S = D * D * D;
    const int lc1 = 31 - __builtin_clz((unsigned)(2 * ch));

    dim3 gu((Nb * Si + 15) / 16, 8, ch / 16);
    upconv_wmma_nhwc<<<gu, 32, 0, stream>>>(xbf, uwb[s], UB[s], catb,
                                            Nb, CinUp, ch, Di, Di, Di, 2 * ch);
    int tot = Nb * ch * S;
    act_to_nhwc<<<(tot + 255) / 256, 256, 0, stream>>>(SK[s], catb, Nb, ch, S, 2 * ch, ch);

    dim3 gc((Nb * S + 127) / 128, ch / 16);
    // conv1: Cin = 2ch (128/64/32) -> always the tap-outer fast path
    size_t smem1 = (size_t)(16 * 27 * 2 * ch) * 2;
    conv3_tap_wmma<<<gc, 256, smem1, stream>>>(catb, wb1[s], B1[s], tbuf, nullptr,
                                               Nb, 2 * ch, lc1, ch, D, D, D);
    // conv2: Cin = ch; fast path for 64/32, generic chunked for 16
    if (ch >= 32) {
      size_t smem2 = (size_t)(16 * 27 * ch) * 2;
      conv3_tap_wmma<<<gc, 256, smem2, stream>>>(tbuf, wb2[s], B2[s], ubuf, Uf[s],
                                                 Nb, ch, lc1 - 1, ch, D, D, D);
    } else {
      size_t smem2 = (size_t)(16 * 27 * ch + 32) * 2;
      conv3_wmma_nhwc<<<gc, 256, smem2, stream>>>(tbuf, wb2[s], B2[s], ubuf, Uf[s],
                                                  Nb, ch, lc1 - 1, ch, D, D, D);
    }
    xbf = ubuf;
    Di = D;
  }

  const int S2 = 64 * 64 * 64;
  head_conv_kernel<<<(Nb * S2 + 255) / 256, 256, 0, stream>>>(ubuf, OW, outp, Nb, 20, S2);
}
